// TransitionDownLayer_56427280335320
// MI455X (gfx1250) — matrix operand order repaired and checked by reading, not compiled, not run on hardware
//
#include <hip/hip_runtime.h>
#include <hip/hip_bf16.h>

#define NPOINT 4096
#define KNN    16
#define CIN    64
#define COUT   128
#define BATCH  4
#define NPTS   16384
#define ROWS   (BATCH * NPOINT * KNN)   // 262144 GEMM rows
#define BN_EPS 1e-5f

typedef __attribute__((ext_vector_type(16))) __bf16 v16bf;
typedef __attribute__((ext_vector_type(8)))  float  v8f;
typedef __attribute__((ext_vector_type(4)))  unsigned int v4u;
typedef __attribute__((ext_vector_type(8)))  unsigned int v8u;

// ---------------------------------------------------------------------------
// TDM: DMA `ndwords` dwords from global memory into LDS at byte offset
// `lds_off`, per CDNA5 ISA D# layout. 1-row 2D tile,
// data_size = 4 bytes, no multicast / padding / iterate. Issue is wave-level;
// caller gates to one wave and follows with s_wait_tensorcnt + barrier.
// ---------------------------------------------------------------------------
__device__ __forceinline__ void tdm_load_to_lds(unsigned lds_off,
                                                const void* gptr,
                                                unsigned ndwords) {
    unsigned long long ga = (unsigned long long)(uintptr_t)gptr;
    v4u g0;
    g0[0] = 1u;                                   // count=1, user descriptor
    g0[1] = lds_off;                              // lds_addr (bytes)
    g0[2] = (unsigned)ga;                         // global_addr[31:0]
    g0[3] = (unsigned)(ga >> 32) | (2u << 30);    // global_addr[56:32] | type=2
    v8u g1;
    g1[0] = 2u << 16;                             // workgroup_mask=0, data_size=4B
    g1[1] = (ndwords & 0xFFFFu) << 16;            // tensor_dim0[15:0]
    g1[2] = (ndwords >> 16) | (1u << 16);         // tensor_dim0[31:16] | tensor_dim1=1
    g1[3] = (ndwords & 0xFFFFu) << 16;            // tile_dim0
    g1[4] = 1u;                                   // tile_dim1=1, tile_dim2=0
    g1[5] = ndwords;                              // tensor_dim0_stride[31:0]
    g1[6] = 0u;                                   // stride hi / dim1_stride lo
    g1[7] = 0u;
    asm volatile("tensor_load_to_lds %0, %1" :: "s"(g0), "s"(g1) : "memory");
}

// ---------------------------------------------------------------------------
// Kernel 0: zero the BN statistic accumulators (must happen every call).
// ---------------------------------------------------------------------------
__global__ void init_stats_kernel(float* stats /* 4*COUT floats */) {
    int i = blockIdx.x * blockDim.x + threadIdx.x;
    if (i < 4 * COUT) stats[i] = 0.0f;
}

// ---------------------------------------------------------------------------
// Kernel 1: farthest point sampling. One workgroup (1024 threads) per batch.
// Each thread owns 16 points (strided) fully in registers; per step: dist
// update vs centroid, wave32 shfl argmax, 32-wave LDS argmax. 3 barriers/step.
// ---------------------------------------------------------------------------
__global__ __launch_bounds__(1024)
void fps_kernel(const float* __restrict__ xyz,
                int* __restrict__ fps_idx,
                float* __restrict__ new_xyz /* d_out, [B,NPOINT,3] */) {
    const int b = blockIdx.x;
    const int t = threadIdx.x;
    const float* x = xyz + (size_t)b * NPTS * 3;

    float px[16], py[16], pz[16], dist[16];
#pragma unroll
    for (int s = 0; s < 16; ++s) {
        int p = t + (s << 10);
        px[s] = x[p * 3 + 0];
        py[s] = x[p * 3 + 1];
        pz[s] = x[p * 3 + 2];
        dist[s] = 1e10f;
    }

    __shared__ float s_c[3];
    __shared__ float s_val[32];
    __shared__ int   s_idx[32];

    int farthest = 0;
    for (int it = 0; it < NPOINT; ++it) {
        if (t == 0) {
            fps_idx[b * NPOINT + it] = farthest;
            float cx = x[farthest * 3 + 0];
            float cy = x[farthest * 3 + 1];
            float cz = x[farthest * 3 + 2];
            s_c[0] = cx; s_c[1] = cy; s_c[2] = cz;
            float* o = new_xyz + ((size_t)b * NPOINT + it) * 3;
            o[0] = cx; o[1] = cy; o[2] = cz;
        }
        __syncthreads();
        const float cx = s_c[0], cy = s_c[1], cz = s_c[2];

        float bestv = -1.0f;
        int   besti = 0;
#pragma unroll
        for (int s = 0; s < 16; ++s) {
            float dx = px[s] - cx, dy = py[s] - cy, dz = pz[s] - cz;
            float d = dx * dx + dy * dy + dz * dz;
            dist[s] = fminf(dist[s], d);
            int p = t + (s << 10);
            if (dist[s] > bestv || (dist[s] == bestv && p < besti)) {
                bestv = dist[s]; besti = p;
            }
        }
#pragma unroll
        for (int off = 16; off >= 1; off >>= 1) {
            float ov = __shfl_xor(bestv, off, 32);
            int   oi = __shfl_xor(besti, off, 32);
            if (ov > bestv || (ov == bestv && oi < besti)) { bestv = ov; besti = oi; }
        }
        if ((t & 31) == 0) { s_val[t >> 5] = bestv; s_idx[t >> 5] = besti; }
        __syncthreads();
        if (t < 32) {
            float v2 = s_val[t];
            int   i2 = s_idx[t];
#pragma unroll
            for (int off = 16; off >= 1; off >>= 1) {
                float ov = __shfl_xor(v2, off, 32);
                int   oi = __shfl_xor(i2, off, 32);
                if (ov > v2 || (ov == v2 && oi < i2)) { v2 = ov; i2 = oi; }
            }
            if (t == 0) s_idx[0] = i2;
        }
        __syncthreads();
        farthest = s_idx[0];
    }
}

// ---------------------------------------------------------------------------
// Kernel 2: kNN. One thread per sampled point; register top-16 (sorted asc).
// ---------------------------------------------------------------------------
__global__ __launch_bounds__(256)
void knn_kernel(const float* __restrict__ xyz,
                const float* __restrict__ new_xyz,
                int* __restrict__ knn_idx) {
    int q = blockIdx.x * blockDim.x + threadIdx.x;
    if (q >= BATCH * NPOINT) return;
    const int b = q / NPOINT;
    const float* x = xyz + (size_t)b * NPTS * 3;
    const float qx = new_xyz[q * 3 + 0];
    const float qy = new_xyz[q * 3 + 1];
    const float qz = new_xyz[q * 3 + 2];

    float bd[KNN];
    int   bi[KNN];
#pragma unroll
    for (int i = 0; i < KNN; ++i) { bd[i] = 3.0e38f; bi[i] = 0; }

    for (int n = 0; n < NPTS; ++n) {
        float dx = x[n * 3 + 0] - qx;
        float dy = x[n * 3 + 1] - qy;
        float dz = x[n * 3 + 2] - qz;
        float d = dx * dx + dy * dy + dz * dz;
        if (d < bd[KNN - 1]) {
            float cv = d; int ci = n;
#pragma unroll
            for (int i = 0; i < KNN; ++i) {
                if (cv < bd[i]) {
                    float tv = bd[i]; int ti = bi[i];
                    bd[i] = cv; bi[i] = ci;
                    cv = tv; ci = ti;
                }
            }
        }
    }
    int* out = knn_idx + (size_t)q * KNN;
#pragma unroll
    for (int i = 0; i < KNN; ++i) out[i] = bi[i];
}

// ---------------------------------------------------------------------------
// Kernel 3: convert a [COUT, C] f32 weight matrix into the exact per-lane
// WMMA B-fragment layout (bf16): flat half-index
//   i = ((s*8 + j)*32 + lane)*16 + 2v + h
// holds B element (k = s*32 + (lane>>4)*16 + 2v + h, n = lane&15) of N-tile j.
// ---------------------------------------------------------------------------
__global__ __launch_bounds__(256)
void prep_weights_kernel(const float* __restrict__ w,
                         __bf16* __restrict__ wbf,
                         int C, int ksteps) {
    int i = blockIdx.x * blockDim.x + threadIdx.x;
    if (i >= ksteps * 8 * 512) return;
    const int h    = i & 1;
    const int v    = (i >> 1) & 7;
    const int lane = (i >> 4) & 31;
    const int j    = (i >> 9) & 7;
    const int s    = i >> 12;
    const int k    = s * 32 + ((lane >> 4) << 4) + (v << 1) + h;
    const int row  = j * 16 + (lane & 15);
    wbf[i] = (__bf16)w[(size_t)row * C + k];
}

// ---------------------------------------------------------------------------
// Kernel 4: gather + GEMM1 (y1 = gather(features) @ w1^T + b1), bf16 WMMA.
// Weights (16 KB, WMMA layout) are TDM-DMA'd into LDS once per block; waves
// read B fragments as contiguous 32 B ds loads. Per-channel BN1 stats via
// LDS ds_add + one global atomic per channel per block.
// ---------------------------------------------------------------------------
__global__ __launch_bounds__(256)
void gemm1_kernel(const float* __restrict__ feat,
                  const __bf16* __restrict__ w1bf,
                  const float* __restrict__ b1,
                  const int* __restrict__ knn_idx,
                  __bf16* __restrict__ y1,
                  float* __restrict__ sum1,
                  float* __restrict__ sq1) {
    __shared__ v16bf wlds[2 * 8 * 32];            // 16 KB
    __shared__ float ls[COUT], lq[COUT];
    const int t = threadIdx.x;
    if (t < COUT) { ls[t] = 0.0f; lq[t] = 0.0f; }
    if (t < 32) {
        tdm_load_to_lds((unsigned)(uintptr_t)&wlds[0], w1bf, 2 * 8 * 32 * 8);
        __builtin_amdgcn_s_wait_tensorcnt(0);
    }
    __syncthreads();

    const int wave = t >> 5, lane = t & 31;
    const int rowTile = blockIdx.x * 8 + wave;    // 16384 tiles total
    const int n = lane & 15;                      // == A-row m for this lane
    const int half = lane >> 4;

    const int r = rowTile * 16 + n;               // global GEMM row
    const int b = r >> 16;                        // 65536 rows per batch
    const int p = knn_idx[r];
    const float* frow = feat + ((size_t)b * NPTS + p) * CIN;

    v8f acc[8];
#pragma unroll
    for (int j = 0; j < 8; ++j) {
        const float bias = b1[j * 16 + n];
#pragma unroll
        for (int v = 0; v < 8; ++v) acc[j][v] = bias;
    }

#pragma unroll
    for (int s = 0; s < 2; ++s) {                 // K = 64 -> 2 steps of 32
        v16bf A;
#pragma unroll
        for (int j = 0; j < 8; ++j) {
            const int c = s * 32 + (half << 4) + (j << 1);
            A[2 * j]     = (__bf16)frow[c];
            A[2 * j + 1] = (__bf16)frow[c + 1];
        }
#pragma unroll
        for (int j = 0; j < 8; ++j) {
            v16bf Bm = wlds[(s * 8 + j) * 32 + lane];
            acc[j] = __builtin_amdgcn_wmma_f32_16x16x32_bf16(
                false, A, false, Bm, (short)0, acc[j], false, false);
        }
    }

#pragma unroll
    for (int j = 0; j < 8; ++j) {
        const int ch = j * 16 + n;
        float s0 = 0.0f, s1 = 0.0f;
#pragma unroll
        for (int v = 0; v < 8; ++v) {
            const float val = acc[j][v];
            s0 += val; s1 += val * val;
            const int row = rowTile * 16 + (half << 3) + v;   // D: M = v + 8*half
            y1[(size_t)row * COUT + ch] = (__bf16)val;
        }
        atomicAdd(&ls[ch], s0);
        atomicAdd(&lq[ch], s1);
    }
    __syncthreads();
    if (t < COUT)            atomicAdd(&sum1[t], ls[t]);
    else if (t < 2 * COUT)   atomicAdd(&sq1[t - COUT], lq[t - COUT]);
}

// ---------------------------------------------------------------------------
// Kernel 5: BN finalize -> per-channel scale/shift (biased variance).
// ---------------------------------------------------------------------------
__global__ void bn_finalize_kernel(const float* __restrict__ sum,
                                   const float* __restrict__ sq,
                                   const float* __restrict__ g,
                                   const float* __restrict__ be,
                                   float* __restrict__ scale,
                                   float* __restrict__ shift) {
    int c = threadIdx.x;
    if (c >= COUT) return;
    const float inv = 1.0f / (float)ROWS;
    const float mean = sum[c] * inv;
    const float var = sq[c] * inv - mean * mean;
    const float sc = g[c] * rsqrtf(var + BN_EPS);
    scale[c] = sc;
    shift[c] = be[c] - mean * sc;
}

// ---------------------------------------------------------------------------
// Kernel 6: GEMM2 (y2 = relu(bn1(y1)) @ w2^T + b2). K = 128 -> 4 steps of 32.
// W2 (32 KB, WMMA layout) TDM-DMA'd to LDS once per block. BN1+ReLU applied
// in-register on the bf16 A loads; BN2 stats like GEMM1.
// ---------------------------------------------------------------------------
__global__ __launch_bounds__(256)
void gemm2_kernel(const __bf16* __restrict__ y1,
                  const __bf16* __restrict__ w2bf,
                  const float* __restrict__ b2,
                  const float* __restrict__ scale1,
                  const float* __restrict__ shift1,
                  __bf16* __restrict__ y2,
                  float* __restrict__ sum2,
                  float* __restrict__ sq2) {
    __shared__ v16bf wlds[4 * 8 * 32];            // 32 KB
    __shared__ float ls[COUT], lq[COUT];
    const int t = threadIdx.x;
    if (t < COUT) { ls[t] = 0.0f; lq[t] = 0.0f; }
    if (t < 32) {
        tdm_load_to_lds((unsigned)(uintptr_t)&wlds[0], w2bf, 4 * 8 * 32 * 8);
        __builtin_amdgcn_s_wait_tensorcnt(0);
    }
    __syncthreads();

    const int wave = t >> 5, lane = t & 31;
    const int rowTile = blockIdx.x * 8 + wave;
    const int n = lane & 15;
    const int half = lane >> 4;

    const int r = rowTile * 16 + n;
    const __bf16* yrow = y1 + (size_t)r * COUT;

    v8f acc[8];
#pragma unroll
    for (int j = 0; j < 8; ++j) {
        const float bias = b2[j * 16 + n];
#pragma unroll
        for (int v = 0; v < 8; ++v) acc[j][v] = bias;
    }

#pragma unroll
    for (int s = 0; s < 4; ++s) {                 // K = 128 -> 4 steps of 32
        v16bf A;
#pragma unroll
        for (int j = 0; j < 8; ++j) {
            const int c = s * 32 + (half << 4) + (j << 1);
            float a0 = fmaxf((float)yrow[c]     * scale1[c]     + shift1[c],     0.0f);
            float a1 = fmaxf((float)yrow[c + 1] * scale1[c + 1] + shift1[c + 1], 0.0f);
            A[2 * j]     = (__bf16)a0;
            A[2 * j + 1] = (__bf16)a1;
        }
#pragma unroll
        for (int j = 0; j < 8; ++j) {
            v16bf Bm = wlds[(s * 8 + j) * 32 + lane];
            acc[j] = __builtin_amdgcn_wmma_f32_16x16x32_bf16(
                false, A, false, Bm, (short)0, acc[j], false, false);
        }
    }

#pragma unroll
    for (int j = 0; j < 8; ++j) {
        const int ch = j * 16 + n;
        float s0 = 0.0f, s1 = 0.0f;
#pragma unroll
        for (int v = 0; v < 8; ++v) {
            const float val = acc[j][v];
            s0 += val; s1 += val * val;
            const int row = rowTile * 16 + (half << 3) + v;
            y2[(size_t)row * COUT + ch] = (__bf16)val;
        }
        atomicAdd(&ls[ch], s0);
        atomicAdd(&lq[ch], s1);
    }
    __syncthreads();
    if (t < COUT)            atomicAdd(&sum2[t], ls[t]);
    else if (t < 2 * COUT)   atomicAdd(&sq2[t - COUT], lq[t - COUT]);
}

// ---------------------------------------------------------------------------
// Kernel 7: apply BN2 + ReLU + max-pool over K -> d_out feature block.
// ---------------------------------------------------------------------------
__global__ __launch_bounds__(256)
void pool_kernel(const __bf16* __restrict__ y2,
                 const float* __restrict__ scale2,
                 const float* __restrict__ shift2,
                 float* __restrict__ out_feat) {
    int i = blockIdx.x * blockDim.x + threadIdx.x;
    if (i >= BATCH * NPOINT * COUT) return;
    const int c = i & (COUT - 1);
    const int q = i >> 7;                          // b*NPOINT + m
    const __bf16* base = y2 + ((size_t)q * KNN) * COUT + c;
    const float sc = scale2[c], sh = shift2[c];
    float mx = 0.0f;                               // relu floor
#pragma unroll
    for (int k = 0; k < KNN; ++k) {
        const float v = (float)base[(size_t)k * COUT] * sc + sh;
        mx = fmaxf(mx, v);
    }
    out_feat[i] = mx;
}

// ---------------------------------------------------------------------------
// Host launcher.
// ---------------------------------------------------------------------------
extern "C" void kernel_launch(void* const* d_in, const int* in_sizes, int n_in,
                              void* d_out, int out_size, void* d_ws, size_t ws_size,
                              hipStream_t stream) {
    const float* xyz  = (const float*)d_in[0];
    const float* feat = (const float*)d_in[1];
    const float* w1   = (const float*)d_in[2];
    const float* b1   = (const float*)d_in[3];
    const float* g1   = (const float*)d_in[4];
    const float* be1  = (const float*)d_in[5];
    const float* w2   = (const float*)d_in[6];
    const float* b2   = (const float*)d_in[7];
    const float* g2   = (const float*)d_in[8];
    const float* be2  = (const float*)d_in[9];

    float* new_xyz  = (float*)d_out;                              // [B,NPOINT,3]
    float* out_feat = (float*)d_out + (size_t)BATCH * NPOINT * 3; // [B,NPOINT,COUT]

    // workspace carve-out (256B aligned)
    char* ws = (char*)d_ws;
    size_t off = 0;
    auto carve = [&](size_t bytes) {
        void* p = ws + off;
        off += (bytes + 255) & ~(size_t)255;
        return p;
    };
    int*    fps_idx = (int*)carve((size_t)BATCH * NPOINT * sizeof(int));
    int*    knn_idx = (int*)carve((size_t)ROWS * sizeof(int));
    __bf16* y1      = (__bf16*)carve((size_t)ROWS * COUT * sizeof(__bf16));
    __bf16* y2      = (__bf16*)carve((size_t)ROWS * COUT * sizeof(__bf16));
    __bf16* w1bf    = (__bf16*)carve((size_t)2 * 8 * 512 * sizeof(__bf16)); // 16 KB
    __bf16* w2bf    = (__bf16*)carve((size_t)4 * 8 * 512 * sizeof(__bf16)); // 32 KB
    float*  stats   = (float*)carve(4 * COUT * sizeof(float));   // sum1,sq1,sum2,sq2
    float*  sum1 = stats, *sq1 = stats + COUT, *sum2 = stats + 2 * COUT, *sq2 = stats + 3 * COUT;
    float*  scale1 = (float*)carve(COUT * sizeof(float));
    float*  shift1 = (float*)carve(COUT * sizeof(float));
    float*  scale2 = (float*)carve(COUT * sizeof(float));
    float*  shift2 = (float*)carve(COUT * sizeof(float));
    (void)ws_size; (void)in_sizes; (void)n_in; (void)out_size;

    init_stats_kernel<<<2, 256, 0, stream>>>(stats);
    prep_weights_kernel<<<(2 * 8 * 512 + 255) / 256, 256, 0, stream>>>(w1, w1bf, CIN, 2);
    prep_weights_kernel<<<(4 * 8 * 512 + 255) / 256, 256, 0, stream>>>(w2, w2bf, COUT, 4);
    fps_kernel<<<BATCH, 1024, 0, stream>>>(xyz, fps_idx, new_xyz);
    knn_kernel<<<(BATCH * NPOINT + 255) / 256, 256, 0, stream>>>(xyz, new_xyz, knn_idx);
    gemm1_kernel<<<ROWS / 16 / 8, 256, 0, stream>>>(feat, w1bf, b1, knn_idx, y1, sum1, sq1);
    bn_finalize_kernel<<<1, COUT, 0, stream>>>(sum1, sq1, g1, be1, scale1, shift1);
    gemm2_kernel<<<ROWS / 16 / 8, 256, 0, stream>>>(y1, w2bf, b2, scale1, shift1, y2, sum2, sq2);
    bn_finalize_kernel<<<1, COUT, 0, stream>>>(sum2, sq2, g2, be2, scale2, shift2);
    pool_kernel<<<((size_t)BATCH * NPOINT * COUT + 255) / 256, 256, 0, stream>>>(y2, scale2, shift2, out_feat);
}
